// ImprovedDyWAN_NDE_80985903333582
// MI455X (gfx1250) — matrix operand order repaired and checked by reading, not compiled, not run on hardware
//
#include <hip/hip_runtime.h>
#include <cstdint>
#include <cstddef>

typedef __attribute__((ext_vector_type(16))) _Float16 v16h;
typedef __attribute__((ext_vector_type(8)))  _Float16 v8h;
typedef __attribute__((ext_vector_type(8)))  float    v8f;

static constexpr int B_   = 8;
static constexpr int C_   = 64;
static constexpr int L_   = 2048;
static constexpr int CS_  = 256;                 // (1+level)*C
static constexpr int NEL  = B_ * L_ * CS_;       // 4,194,304
static constexpr int NELC = B_ * L_ * C_;        // 1,048,576

__device__ __forceinline__ float gelu_f(float x) {
  return 0.5f * x * (1.0f + erff(x * 0.70710678118654752f));
}

// ---------------------------------------------------------------------------
// Implicit-im2col conv1d (k=3, pad=1) as WMMA GEMM.
//   D(16x16) = A(Mtile=16 x K32) x B(K32 x Ntile=16) accumulated over K chunks.
//   A = weights, prepacked in exact wave32 A-fragment layout (contig per lane).
//   B = activations, staged to LDS in [pos][cin] f16 with padded pitch.
// EPI: 1 = bias+FiLM+GELU -> f16 NLC, 2 = bias+GELU -> f16 NLC, 3 = bias -> f32 NLC
// ---------------------------------------------------------------------------
template <int CIN, int COUT, int EPI>
__global__ __launch_bounds__(128)
void convgemm_k(const _Float16* __restrict__ act16,
                const _Float16* __restrict__ apack,
                const float*    __restrict__ bias,
                const float*    __restrict__ gb,     // gamma[0:64], beta[64:128] (EPI==1)
                _Float16*       __restrict__ outh,   // EPI 1/2
                float*          __restrict__ outf)   // EPI 3
{
  constexpr int NP    = 64;             // positions per workgroup (4 waves x 16)
  constexpr int NKC   = (3 * CIN) / 32; // K chunks of 32
  constexpr int NMT   = COUT / 16;      // M tiles
  constexpr int PITCH = CIN + 8;        // padded LDS row pitch (halves)

  __shared__ _Float16 lds[(NP + 2) * PITCH];

  const int b    = blockIdx.y;
  const int p0   = blockIdx.x * NP;
  const int tid  = threadIdx.x;
  const int lane = tid & 31;
  const int wv   = tid >> 5;

  // Stage activation slab rows p0-1 .. p0+NP (zero pad at batch edges).
  for (int v = tid; v < (NP + 2) * (CIN / 8); v += 128) {
    const int row = v / (CIN / 8);
    const int col = (v % (CIN / 8)) * 8;
    const int p   = p0 + row - 1;
    v8h val = {};
    if (p >= 0 && p < L_)
      val = *(const v8h*)&act16[(size_t(b) * L_ + p) * CIN + col];
    *(v8h*)&lds[row * PITCH + col] = val;
  }
  __syncthreads();

  const int n    = lane & 15;
  const int hi   = lane >> 4;
  const int nrow = wv * 16 + n;           // slab row (before tap shift)
  const int p    = p0 + wv * 16 + n;      // global position this lane's column

  for (int mt = 0; mt < NMT; ++mt) {
    v8f acc = {};
#pragma unroll 2
    for (int kc = 0; kc < NKC; ++kc) {
      const int tap = kc / (CIN / 32);
      const int c0  = (kc % (CIN / 32)) * 32 + hi * 8;
      // A fragment: 16 consecutive halves per lane (prepacked layout).
      v16h av = *(const v16h*)&apack[(size_t(mt) * NKC + kc) * 512 + lane * 16];
      // B fragment: lane(l) holds col n=l&15; element e -> K = (e<8?0:16)+hi*8+(e&7)
      //             -> two runs of 8 input channels at fixed position.
      const _Float16* bp = &lds[(nrow + tap) * PITCH + c0];
      v16h bv;
      *((v8h*)&bv)     = *(const v8h*)(bp);
      *((v8h*)&bv + 1) = *(const v8h*)(bp + 16);
      acc = __builtin_amdgcn_wmma_f32_16x16x32_f16(false, av, false, bv,
                                                   (short)0, acc, false, false);
    }
    // C/D layout: vgpr j -> M = mt*16 + hi*8 + j (8 consecutive rows), N = n.
    const int mb = mt * 16 + hi * 8;
    if constexpr (EPI == 1) {
      v8h hv;
#pragma unroll
      for (int j = 0; j < 8; ++j) {
        const int m = mb + j;
        float vv = acc[j] + bias[m];
        vv = (1.0f + gb[m]) * vv + gb[64 + m];
        hv[j] = (_Float16)gelu_f(vv);
      }
      *(v8h*)&outh[(size_t(b) * L_ + p) * COUT + mb] = hv;
    } else if constexpr (EPI == 2) {
      v8h hv;
#pragma unroll
      for (int j = 0; j < 8; ++j)
        hv[j] = (_Float16)gelu_f(acc[j] + bias[mb + j]);
      *(v8h*)&outh[(size_t(b) * L_ + p) * COUT + mb] = hv;
    } else {
      v8f ov;
#pragma unroll
      for (int j = 0; j < 8; ++j)
        ov[j] = acc[j] + bias[mb + j];
      *(v8f*)&outf[(size_t(b) * L_ + p) * COUT + mb] = ov;
    }
  }
}

// --------------------------- weight prepack --------------------------------
// K ordering: K = tap*CIN + ic  (chunk-of-32 never straddles a tap boundary).
// A-fragment per ISA 16-bit A table: lane l -> M = l&15; element e:
//   K_local = (e<8 ? 0 : 16) + (l>>4)*8 + (e&7)
__global__ void pack_k(const float* __restrict__ k1, const float* __restrict__ k2,
                       const float* __restrict__ k3,
                       _Float16* __restrict__ a1, _Float16* __restrict__ a2,
                       _Float16* __restrict__ a3)
{
  const int idx = blockIdx.x * 256 + threadIdx.x;
  constexpr int N1 = 4 * 24 * 512, N2 = 4 * 6 * 512, N3 = 16 * 6 * 512;
  const float* w; _Float16* out; int cin, nkc, local;
  if (idx < N1)            { w = k1; out = a1; cin = 256; nkc = 24; local = idx; }
  else if (idx < N1 + N2)  { w = k2; out = a2; cin = 64;  nkc = 6;  local = idx - N1; }
  else if (idx < N1 + N2 + N3) { w = k3; out = a3; cin = 64; nkc = 6; local = idx - N1 - N2; }
  else return;
  const int e    = local & 15;
  const int lane = (local >> 4) & 31;
  const int frag = local >> 9;
  const int kc   = frag % nkc;
  const int mt   = frag / nkc;
  const int m    = mt * 16 + (lane & 15);
  const int hi   = lane >> 4;
  const int kl   = ((e < 8) ? 0 : 16) + hi * 8 + (e & 7);
  const int K    = kc * 32 + kl;
  const int tap  = K / cin;
  const int ic   = K % cin;
  out[local] = (_Float16)w[(m * cin + ic) * 3 + tap];
}

// ----------------------- time-embedding table ------------------------------
__global__ void ttab_k(const float* __restrict__ tw1, const float* __restrict__ tb1,
                       const float* __restrict__ tw2, const float* __restrict__ tb2,
                       const float* __restrict__ cw,  const float* __restrict__ cb,
                       float* __restrict__ gbtab)
{
  __shared__ float te[16], te2[16];
  const int s = blockIdx.x;          // 0..24 (24 == t=1.0 via i=4,j=0)
  const int t = threadIdx.x;         // 128
  const int i = s / 6, j = s % 6;
  float off;
  switch (j) {
    case 0: off = 0.0f; break;
    case 1: off = 0.2f; break;
    case 2: off = 0.3f; break;
    case 3: off = 0.8f; break;
    case 4: off = 8.0f / 9.0f; break;
    default: off = 1.0f; break;
  }
  const float tv = i * 0.25f + 0.25f * off;
  if (t < 16) te[t] = gelu_f(tv * tw1[t] + tb1[t]);
  __syncthreads();
  if (t < 16) {
    float s2 = tb2[t];
    for (int k = 0; k < 16; ++k) s2 += te[k] * tw2[t * 16 + k];
    te2[t] = s2;
  }
  __syncthreads();
  float sum = cb[t];
  for (int k = 0; k < 16; ++k) sum += te2[k] * cw[t * 16 + k];
  gbtab[s * 128 + t] = sum;
}

// ----------------------------- DyWAN ---------------------------------------
__global__ void xin_k(const float* __restrict__ x, float* __restrict__ ap,
                      float* __restrict__ stk)
{
  const int idx = blockIdx.x * 256 + threadIdx.x;        // NELC, c fastest
  const int c = idx & 63, l = (idx >> 6) & 2047, b = idx >> 17;
  const float v = x[(size_t(b) * 64 + c) * L_ + l];
  ap[(size_t(b) * L_ + l) * 64 + c]   = v;
  stk[(size_t(b) * L_ + l) * 256 + c] = v;
}

__global__ void rmax_k(const float* __restrict__ ap, float* __restrict__ amax)
{
  const int b = blockIdx.x, c = threadIdx.x;             // 8 x 64
  float m = -3.4e38f;
  for (int l = 0; l < L_; ++l)
    m = fmaxf(m, ap[(size_t(b) * L_ + l) * 64 + c]);
  amax[b * 64 + c] = m;
}

__global__ void dywan_mlp_k(const float* __restrict__ amax,
                            const float* __restrict__ sw1, const float* __restrict__ sb1,
                            const float* __restrict__ sw2, const float* __restrict__ sb2,
                            const float* __restrict__ sw3, const float* __restrict__ sb3,
                            float* __restrict__ lohi)
{
  __shared__ float feat[8 * 32], g[8 * 32], raw[8 * 16], rn[8 * 16];
  const int t = threadIdx.x;                              // 256
  { const int b = t >> 5, h = t & 31;
    float s = sb1[h];
    for (int k = 0; k < 64; ++k) s += amax[b * 64 + k] * sw1[h * 64 + k];
    feat[b * 32 + h] = gelu_f(s); }
  __syncthreads();
  { const int b = t >> 5, h = t & 31;
    float s = sb2[h];
    for (int k = 0; k < 32; ++k) s += feat[b * 32 + k] * sw2[h * 32 + k];
    g[b * 32 + h] = gelu_f(s); }
  __syncthreads();
  if (t < 128) {
    const int b = t >> 4, o = t & 15;
    float s = sb3[o];
    for (int k = 0; k < 32; ++k) s += g[b * 32 + k] * sw3[o * 32 + k];
    raw[b * 16 + o] = s;
  }
  __syncthreads();
  if (t < 16) {
    const int b = t >> 1, half = t & 1;
    float s = 0.f;
    for (int k = 0; k < 8; ++k) { const float v = raw[b * 16 + half * 8 + k]; s += v * v; }
    const float inv = 1.0f / sqrtf(s);
    for (int k = 0; k < 8; ++k) rn[b * 16 + half * 8 + k] = raw[b * 16 + half * 8 + k] * inv;
  }
  __syncthreads();
  if (t < 16) {
    float s = 0.f;
    for (int b = 0; b < 8; ++b) s += rn[b * 16 + t];
    lohi[t] = s * 0.125f;                                  // lohi[0:8]=lo, [8:16]=hi
  }
}

// depthwise reflect-pad(3,4) conv, FL=8; writes lo->next approx, hi->stacked group
__global__ void gconv_k(const float* __restrict__ ap, const float* __restrict__ lohi,
                        float* __restrict__ nap, float* __restrict__ stk, int group)
{
  const int idx = blockIdx.x * 256 + threadIdx.x;          // NELC, c fastest
  const int c = idx & 63, l = (idx >> 6) & 2047, b = idx >> 17;
  float lo = 0.f, hi = 0.f;
#pragma unroll
  for (int k = 0; k < 8; ++k) {
    const int j = l + k;                                   // padded index, 0..2054
    int src;
    if (j < 3)            src = 3 - j;                     // left reflect
    else if (j < L_ + 3)  src = j - 3;                     // body
    else                  src = 2 * L_ + 1 - j;            // right reflect
    const float v = ap[(size_t(b) * L_ + src) * 64 + c];
    lo = fmaf(v, lohi[k], lo);
    hi = fmaf(v, lohi[8 + k], hi);
  }
  nap[(size_t(b) * L_ + l) * 64 + c] = lo;
  stk[(size_t(b) * L_ + l) * 256 + group * 64 + c] = hi;
}

// --------------------------- RK plumbing -----------------------------------
__global__ void stage_input_k(const float* __restrict__ y,
                              const float* __restrict__ a, const float* __restrict__ b2,
                              const float* __restrict__ c, const float* __restrict__ d,
                              const float* __restrict__ e,
                              float ca, float cb, float cc, float cd, float ce, int ns,
                              _Float16* __restrict__ out)
{
  const int i = blockIdx.x * 256 + threadIdx.x;
  float v = y[i];
  if (ns > 0) v = fmaf(ca, a[i], v);
  if (ns > 1) v = fmaf(cb, b2[i], v);
  if (ns > 2) v = fmaf(cc, c[i], v);
  if (ns > 3) v = fmaf(cd, d[i], v);
  if (ns > 4) v = fmaf(ce, e[i], v);
  out[i] = (_Float16)v;
}

__global__ void y_update_k(float* __restrict__ y,
                           const float* __restrict__ s1, const float* __restrict__ s3,
                           const float* __restrict__ s4, const float* __restrict__ s5,
                           const float* __restrict__ s6,
                           float b1, float b3, float b4, float b5, float b6)
{
  const int i = blockIdx.x * 256 + threadIdx.x;
  float v = y[i];
  v = fmaf(b1, s1[i], v);
  v = fmaf(b3, s3[i], v);
  v = fmaf(b4, s4[i], v);
  v = fmaf(b5, s5[i], v);
  v = fmaf(b6, s6[i], v);
  y[i] = v;
}

// ------------------------------ outputs ------------------------------------
__global__ void zero_inner_k(float* __restrict__ inner)
{
  if (threadIdx.x < B_) inner[threadIdx.x] = 0.f;
}

__global__ void dot_reduce_k(const float* __restrict__ st, const float* __restrict__ dx,
                             float* __restrict__ inner)
{
  __shared__ float red[256];
  const int b = blockIdx.y;
  const size_t base = (size_t)b * (L_ * CS_);
  float s = 0.f;
  for (int i = blockIdx.x * 256 + threadIdx.x; i < L_ * CS_; i += 64 * 256)
    s += st[base + i] * dx[base + i];
  red[threadIdx.x] = s;
  __syncthreads();
  for (int k = 128; k > 0; k >>= 1) {
    if (threadIdx.x < k) red[threadIdx.x] += red[threadIdx.x + k];
    __syncthreads();
  }
  if (threadIdx.x == 0) atomicAdd(&inner[b], red[0]);
}

__global__ void ecloss_k(const float* __restrict__ inner, float* __restrict__ out)
{
  if (threadIdx.x == 0) {
    float s = 0.f;
    for (int b = 0; b < B_; ++b) s += inner[b] * inner[b];
    out[0] = s * 0.125f;
  }
}

// NLC y -> (yl,yh0,yh1,yh2) NCH concat
__global__ void final_out_k(const float* __restrict__ y, float* __restrict__ dout)
{
  const int idx = blockIdx.x * 256 + threadIdx.x;          // NEL, l fastest
  const int l = idx & 2047, c = (idx >> 11) & 255, b = idx >> 19;
  const int g = c >> 6, co = c & 63;
  dout[size_t(g) * (B_ * 64 * L_) + (size_t(b) * 64 + co) * L_ + l] =
      y[(size_t(b) * L_ + l) * 256 + c];
}

// ---------------------------------------------------------------------------
extern "C" void kernel_launch(void* const* d_in, const int* in_sizes, int n_in,
                              void* d_out, int out_size, void* d_ws, size_t ws_size,
                              hipStream_t stream)
{
  (void)in_sizes; (void)n_in; (void)ws_size;
  const float* x   = (const float*)d_in[0];
  const float* sw1 = (const float*)d_in[1];
  const float* sb1 = (const float*)d_in[2];
  const float* sw2 = (const float*)d_in[3];
  const float* sb2 = (const float*)d_in[4];
  const float* sw3 = (const float*)d_in[5];
  const float* sb3 = (const float*)d_in[6];
  const float* tw1 = (const float*)d_in[7];
  const float* tb1 = (const float*)d_in[8];
  const float* tw2 = (const float*)d_in[9];
  const float* tb2 = (const float*)d_in[10];
  const float* cw  = (const float*)d_in[11];
  const float* cbv = (const float*)d_in[12];
  const float* k1  = (const float*)d_in[13];
  const float* kb1 = (const float*)d_in[14];
  const float* k2  = (const float*)d_in[15];
  const float* kb2 = (const float*)d_in[16];
  const float* k3  = (const float*)d_in[17];
  const float* kb3 = (const float*)d_in[18];
  float* dout = (float*)d_out;

  char* wp = (char*)d_ws;
  auto alloc = [&](size_t bytes) -> char* {
    char* p = wp;
    wp += (bytes + 255) & ~(size_t)255;
    return p;
  };
  float* stacked = (float*)alloc(sizeof(float) * NEL);
  float* y       = (float*)alloc(sizeof(float) * NEL);
  float* S1      = (float*)alloc(sizeof(float) * NEL);
  float* S2      = (float*)alloc(sizeof(float) * NEL);
  float* S3      = (float*)alloc(sizeof(float) * NEL);
  float* S4      = (float*)alloc(sizeof(float) * NEL);
  float* S5      = (float*)alloc(sizeof(float) * NEL);
  float* S6      = S2;   // safe alias: s2 last read when building s6's input
  _Float16* ys16 = (_Float16*)alloc(sizeof(_Float16) * NEL);
  _Float16* h16  = (_Float16*)alloc(sizeof(_Float16) * NELC);
  _Float16* h216 = (_Float16*)alloc(sizeof(_Float16) * NELC);
  _Float16* a1   = (_Float16*)alloc(sizeof(_Float16) * (4 * 24 * 512));
  _Float16* a2   = (_Float16*)alloc(sizeof(_Float16) * (4 * 6 * 512));
  _Float16* a3   = (_Float16*)alloc(sizeof(_Float16) * (16 * 6 * 512));
  float* gbtab   = (float*)alloc(sizeof(float) * 25 * 128);
  float* amax    = (float*)alloc(sizeof(float) * B_ * C_);
  float* lohi    = (float*)alloc(sizeof(float) * 16);
  float* inner   = (float*)alloc(sizeof(float) * B_);
  float* apA = S3;       // pre-ODE scratch (overwritten later by stages)
  float* apB = S4;

  const int TB = 256;

  // ---- DyWAN wavelet front-end ----
  xin_k<<<NELC / TB, TB, 0, stream>>>(x, apA, stacked);
  float* cur = apA; float* nxt = apB;
  for (int lev = 0; lev < 3; ++lev) {
    rmax_k<<<B_, C_, 0, stream>>>(cur, amax);
    dywan_mlp_k<<<1, 256, 0, stream>>>(amax, sw1, sb1, sw2, sb2, sw3, sb3, lohi);
    gconv_k<<<NELC / TB, TB, 0, stream>>>(cur, lohi, nxt, stacked, lev + 1);
    float* t = cur; cur = nxt; nxt = t;
  }
  hipMemcpyAsync(y, stacked, sizeof(float) * NEL, hipMemcpyDeviceToDevice, stream);

  // ---- one-time per-call prep ----
  ttab_k<<<25, 128, 0, stream>>>(tw1, tb1, tw2, tb2, cw, cbv, gbtab);
  pack_k<<<(110592 + TB - 1) / TB, TB, 0, stream>>>(k1, k2, k3, a1, a2, a3);
  zero_inner_k<<<1, 32, 0, stream>>>(inner);

  const float hs = 0.25f;
  auto SI = [&](const float* A, const float* Bq, const float* Cq, const float* Dq,
                const float* Eq, float ca, float cb, float cc, float cd, float ce, int ns) {
    stage_input_k<<<NEL / TB, TB, 0, stream>>>(y, A, Bq, Cq, Dq, Eq,
                                               ca, cb, cc, cd, ce, ns, ys16);
  };
  auto ODE = [&](int st, float* sout) {
    convgemm_k<256, 64, 1><<<dim3(L_ / 64, B_), 128, 0, stream>>>(
        ys16, a1, kb1, gbtab + st * 128, h16, nullptr);
    convgemm_k<64, 64, 2><<<dim3(L_ / 64, B_), 128, 0, stream>>>(
        h16, a2, kb2, nullptr, h216, nullptr);
    convgemm_k<64, 256, 3><<<dim3(L_ / 64, B_), 128, 0, stream>>>(
        h216, a3, kb3, nullptr, nullptr, sout);
  };

  // ---- Dormand-Prince RK45, 4 fixed steps ----
  for (int i = 0; i < 4; ++i) {
    const int st = i * 6;
    SI(y, y, y, y, y, 0, 0, 0, 0, 0, 0);                                   ODE(st + 0, S1);
    SI(S1, y, y, y, y, hs * (1.f / 5.f), 0, 0, 0, 0, 1);                   ODE(st + 1, S2);
    SI(S1, S2, y, y, y, hs * (3.f / 40.f), hs * (9.f / 40.f), 0, 0, 0, 2); ODE(st + 2, S3);
    SI(S1, S2, S3, y, y, hs * (44.f / 45.f), hs * (-56.f / 15.f),
       hs * (32.f / 9.f), 0, 0, 3);                                        ODE(st + 3, S4);
    SI(S1, S2, S3, S4, y, hs * (19372.f / 6561.f), hs * (-25360.f / 2187.f),
       hs * (64448.f / 6561.f), hs * (-212.f / 729.f), 0, 4);              ODE(st + 4, S5);
    SI(S1, S2, S3, S4, S5, hs * (9017.f / 3168.f), hs * (-355.f / 33.f),
       hs * (46732.f / 5247.f), hs * (49.f / 176.f),
       hs * (-5103.f / 18656.f), 5);                                       ODE(st + 5, S6);
    y_update_k<<<NEL / TB, TB, 0, stream>>>(
        y, S1, S3, S4, S5, S6,
        hs * (35.f / 384.f), hs * (500.f / 1113.f), hs * (125.f / 192.f),
        hs * (-2187.f / 6784.f), hs * (11.f / 84.f));
  }

  // ---- dx = ode_f(1.0, ode_out), ecloss, outputs ----
  SI(y, y, y, y, y, 0, 0, 0, 0, 0, 0);
  ODE(24, S1);
  dot_reduce_k<<<dim3(64, B_), 256, 0, stream>>>(stacked, S1, inner);
  final_out_k<<<NEL / TB, TB, 0, stream>>>(y, dout);
  ecloss_k<<<1, 1, 0, stream>>>(inner, dout + (out_size - 1));
}
